// QuantLinear_75067438400150
// MI455X (gfx1250) — matrix operand order repaired and checked by reading, not compiled, not run on hardware
//
#include <hip/hip_runtime.h>

// ---------------------------------------------------------------------------
// Types
// ---------------------------------------------------------------------------
typedef __attribute__((ext_vector_type(16))) _Float16 v16h;
typedef __attribute__((ext_vector_type(8)))  _Float16 v8h;
typedef __attribute__((ext_vector_type(4)))  _Float16 v4h;
typedef __attribute__((ext_vector_type(8)))  float    v8f;
typedef __attribute__((ext_vector_type(4)))  float    v4f;
typedef __attribute__((ext_vector_type(4)))  int      v4i;

#define FP8_MAX   448.0f
#define EPS_SCALE 1e-8f

// ---------------------------------------------------------------------------
// CDNA5 async global->LDS copy (ASYNCcnt path), with safe fallbacks.
// Builtin signature on this toolchain: (v4i addrspace(1)*, v4i addrspace(3)*,
// imm offset, imm cpol).
// ---------------------------------------------------------------------------
#if __has_builtin(__builtin_amdgcn_global_load_async_to_lds_b128)
#define HAVE_ASYNC_LDS 1
#else
#define HAVE_ASYNC_LDS 0
#endif

typedef __attribute__((address_space(1))) v4i* gptr_v4i;
typedef __attribute__((address_space(3))) v4i* lptr_v4i;

__device__ __forceinline__ void async_copy_b128(const _Float16* g, _Float16* l) {
#if HAVE_ASYNC_LDS
  __builtin_amdgcn_global_load_async_to_lds_b128(
      (gptr_v4i)(void*)g, (lptr_v4i)(void*)l, 0, 0);
#else
  *(uint4*)l = *(const uint4*)g;   // sync fallback through VGPRs
#endif
}

__device__ __forceinline__ void wait_async_all() {
#if HAVE_ASYNC_LDS
#if __has_builtin(__builtin_amdgcn_s_wait_asynccnt)
  __builtin_amdgcn_s_wait_asynccnt(0);
#else
  asm volatile("s_wait_asynccnt 0x0" ::: "memory");
#endif
#endif
}

// ---------------------------------------------------------------------------
// Software fp8-e4m3fn round-to-nearest-even (input already scaled, |y|<=448)
// ---------------------------------------------------------------------------
__device__ __forceinline__ float e4m3_rtne(float y) {
  float ay = __builtin_fabsf(y);
  if (ay < 0.015625f) {
    // subnormal range of e4m3 (<2^-6): step 2^-9. magic = 1.5*2^14 keeps the
    // sum in [2^14, 2^15) where float ulp == 2^-9 -> RNE for either sign.
    const float magic = 24576.0f;
    float t = y + magic;
    return t - magic;
  }
  // normal range: RNE the float32 mantissa down to 3 bits (drop 20 bits).
  unsigned int u = __float_as_uint(y);
  unsigned int lsb = (u >> 20) & 1u;
  u += 0x7FFFFu + lsb;
  u &= 0xFFF00000u;
  return __uint_as_float(u);
}

// ---------------------------------------------------------------------------
// Groupwise (32 along last dim) fp8 pseudo-quantize fp32 -> dequantized f16.
// Each lane owns 4 consecutive elements; a group of 32 spans 8 lanes, so the
// amax reduction is a width-8 xor-shuffle tree (wave32 native).
// ---------------------------------------------------------------------------
__global__ __launch_bounds__(256) void quant_dq_f16_kernel(
    const float* __restrict__ in, _Float16* __restrict__ out, long nvec4) {
  long i = (long)blockIdx.x * blockDim.x + threadIdx.x;  // float4 index
  if (i >= nvec4) return;

  v4f x = ((const v4f*)in)[i];
  float amax = __builtin_fmaxf(
      __builtin_fmaxf(__builtin_fabsf(x.x), __builtin_fabsf(x.y)),
      __builtin_fmaxf(__builtin_fabsf(x.z), __builtin_fabsf(x.w)));
  amax = __builtin_fmaxf(amax, __shfl_xor(amax, 1, 8));
  amax = __builtin_fmaxf(amax, __shfl_xor(amax, 2, 8));
  amax = __builtin_fmaxf(amax, __shfl_xor(amax, 4, 8));

  float scale = __builtin_fmaxf(amax, EPS_SCALE) * (1.0f / FP8_MAX);
  float inv   = 1.0f / scale;

  v4h q;
  q.x = (_Float16)(e4m3_rtne(x.x * inv) * scale);
  q.y = (_Float16)(e4m3_rtne(x.y * inv) * scale);
  q.z = (_Float16)(e4m3_rtne(x.z * inv) * scale);
  q.w = (_Float16)(e4m3_rtne(x.w * inv) * scale);
  ((v4h*)out)[i] = q;
}

// ---------------------------------------------------------------------------
// Tiled f16 WMMA GEMM:  out[m,n] = sum_k A[m,k]*B[n,k] + bias[n]
//   A: [M,K] f16 row-major, B: [N,K] f16 row-major (W layout, i.e. x @ W^T)
// Block tile 128x128, BK=64, double-buffered LDS filled via async-to-LDS,
// 8 waves arranged 4(M) x 2(N), wave tile 32x64 = 2x4 WMMA accumulators.
// ---------------------------------------------------------------------------
#define BM  128
#define BN  128
#define BK  64
#define LDP 72   // padded LDS row stride in f16 (144B -> conflict-free reads)

__global__ __launch_bounds__(256) void gemm_f16_wmma_kernel(
    const _Float16* __restrict__ Aq,   // [M,K]
    const _Float16* __restrict__ Bq,   // [N,K]
    const float*    __restrict__ bias, // [N]
    float*          __restrict__ out,  // [M,N]
    int M, int N, int K) {
  __shared__ __attribute__((aligned(16))) _Float16 AsBuf[2][BM][LDP];
  __shared__ __attribute__((aligned(16))) _Float16 BsBuf[2][BN][LDP];

  const int tid  = threadIdx.x;
  const int lane = tid & 31;
  const int wave = tid >> 5;     // 0..7
  const int wm   = wave & 3;     // wave row (M): 0..3 -> 32 rows each
  const int wn   = wave >> 2;    // wave col (N): 0..1 -> 64 cols each
  const int half = lane >> 4;    // hi/lo 16-lane half (ISA frag layout)
  const int l16  = lane & 15;

  const int bm = blockIdx.y * BM;
  const int bn = blockIdx.x * BN;

  // cooperative tile load mapping: each thread moves 32 f16 (4 x b128)
  const int lr = tid >> 1;           // tile row 0..127
  const int lc = (tid & 1) * 32;     // tile col base 0 or 32

  const _Float16* aG = Aq + (size_t)(bm + lr) * K + lc;
  const _Float16* bG = Bq + (size_t)(bn + lr) * K + lc;

  v8f acc[2][4];
#pragma unroll
  for (int mi = 0; mi < 2; ++mi)
#pragma unroll
    for (int ni = 0; ni < 4; ++ni)
#pragma unroll
      for (int e = 0; e < 8; ++e) acc[mi][ni][e] = 0.0f;

  const int nTiles = K / BK;

  // prologue: stage tile 0 into buffer 0
  {
    const _Float16* ag = aG;
    const _Float16* bg = bG;
    _Float16* as = &AsBuf[0][lr][lc];
    _Float16* bs = &BsBuf[0][lr][lc];
#pragma unroll
    for (int i = 0; i < 4; ++i) async_copy_b128(ag + 8 * i, as + 8 * i);
#pragma unroll
    for (int i = 0; i < 4; ++i) async_copy_b128(bg + 8 * i, bs + 8 * i);
  }

  for (int t = 0; t < nTiles; ++t) {
    const int buf = t & 1;
    wait_async_all();     // tile t resident in LDS (my stores done)
    __syncthreads();      // everyone's stores done; prev reads of buf^1 done

    if (t + 1 < nTiles) { // stage tile t+1 into the other buffer (no wait)
      const _Float16* ag = aG + (size_t)(t + 1) * BK;
      const _Float16* bg = bG + (size_t)(t + 1) * BK;
      _Float16* as = &AsBuf[buf ^ 1][lr][lc];
      _Float16* bs = &BsBuf[buf ^ 1][lr][lc];
#pragma unroll
      for (int i = 0; i < 4; ++i) async_copy_b128(ag + 8 * i, as + 8 * i);
#pragma unroll
      for (int i = 0; i < 4; ++i) async_copy_b128(bg + 8 * i, bs + 8 * i);
      if (t + 2 < nTiles) {  // cover latency beyond one stage
        __builtin_prefetch(aG + (size_t)(t + 2) * BK, 0, 1);
        __builtin_prefetch(bG + (size_t)(t + 2) * BK, 0, 1);
      }
    }

    // compute on tile t
    const _Float16 (*As)[LDP] = AsBuf[buf];
    const _Float16 (*Bs)[LDP] = BsBuf[buf];
#pragma unroll
    for (int ks = 0; ks < BK; ks += 32) {
      // A fragments (16-bit A 16x32: lanes 0-15 hold K{0..7,16..23},
      // lanes 16-31 hold K{8..15,24..31}, row M = lane%16)
      v16h af[2];
#pragma unroll
      for (int mi = 0; mi < 2; ++mi) {
        const int row = wm * 32 + mi * 16 + l16;
        v8h lo = *(const v8h*)&As[row][ks + half * 8];
        v8h hi = *(const v8h*)&As[row][ks + 16 + half * 8];
#pragma unroll
        for (int e = 0; e < 8; ++e) { af[mi][e] = lo[e]; af[mi][e + 8] = hi[e]; }
      }
      // B fragments (16-bit B 32x16: lanes 0-15 -> N=lane, K{0..7,16..23};
      // lanes 16-31 -> N=lane-16, K{8..15,24..31}). Bs is [N][K]: contiguous reads.
      v16h bf[4];
#pragma unroll
      for (int ni = 0; ni < 4; ++ni) {
        const int row = wn * 64 + ni * 16 + l16;
        v8h lo = *(const v8h*)&Bs[row][ks + half * 8];
        v8h hi = *(const v8h*)&Bs[row][ks + 16 + half * 8];
#pragma unroll
        for (int e = 0; e < 8; ++e) { bf[ni][e] = lo[e]; bf[ni][e + 8] = hi[e]; }
      }
#pragma unroll
      for (int mi = 0; mi < 2; ++mi)
#pragma unroll
        for (int ni = 0; ni < 4; ++ni)
          acc[mi][ni] = __builtin_amdgcn_wmma_f32_16x16x32_f16(
              false, af[mi], false, bf[ni], (short)0, acc[mi][ni], false, false);
    }
  }

  // Epilogue: C layout = VGPR r -> M = r + 8*half, N = lane%16.
#pragma unroll
  for (int ni = 0; ni < 4; ++ni) {
    const int n  = bn + wn * 64 + ni * 16 + l16;
    const float bv = bias[n];
#pragma unroll
    for (int mi = 0; mi < 2; ++mi) {
      const int mb = bm + wm * 32 + mi * 16 + half * 8;
#pragma unroll
      for (int r = 0; r < 8; ++r)
        out[(size_t)(mb + r) * N + n] = acc[mi][ni][r] + bv;
    }
  }
}

// ---------------------------------------------------------------------------
// Launch
// ---------------------------------------------------------------------------
extern "C" void kernel_launch(void* const* d_in, const int* in_sizes, int n_in,
                              void* d_out, int out_size, void* d_ws, size_t ws_size,
                              hipStream_t stream) {
  const float* x    = (const float*)d_in[0];  // [B,S,K] = [M,K]
  const float* W    = (const float*)d_in[1];  // [N,K]
  const float* bias = (const float*)d_in[2];  // [N]
  float* out        = (float*)d_out;

  const int N = in_sizes[2];
  const int K = in_sizes[1] / N;
  const int M = in_sizes[0] / K;

  _Float16* xq = (_Float16*)d_ws;
  _Float16* wq = xq + (size_t)M * K;

  // Stage 1: groupwise fp8 pseudo-quant -> dequantized f16 in workspace
  {
    long nvX = (long)M * K / 4;
    long nvW = (long)N * K / 4;
    quant_dq_f16_kernel<<<(unsigned)((nvX + 255) / 256), 256, 0, stream>>>(x, xq, nvX);
    quant_dq_f16_kernel<<<(unsigned)((nvW + 255) / 256), 256, 0, stream>>>(W, wq, nvW);
  }

  // Stage 2: f16 WMMA GEMM (x @ W^T + bias); xq+wq (96MB f16) live in 192MB L2
  {
    dim3 grid((unsigned)(N / BN), (unsigned)(M / BM));
    gemm_f16_wmma_kernel<<<grid, 256, 0, stream>>>(xq, wq, bias, out, M, N, K);
  }
}
